// SpectralConv1d_6244882448387
// MI455X (gfx1250) — matrix operand order repaired and checked by reading, not compiled, not run on hardware
//
#include <hip/hip_runtime.h>
#include <hip/hip_bf16.h>
#include <stdint.h>

// ---------------------------------------------------------------------------
// SpectralConv1d on MI455X (gfx1250):
//   stage0: twiddle tables (f16 hi/lo split, WMMA-B fragment order)
//   stage1: forward partial DFT  GEMM 4096x128x8192
//           TDM -> LDS f32 -> cooperative hi/lo split (fragment order in LDS)
//           -> ds_load_b128 fragments -> WMMA f16 (3-term split product)
//   stage2: complex mode mixing + irfft coefficient packing (f32 exact)
//   stage3: inverse partial DFT  GEMM 4096x8192x128 (fragment-order loads -> WMMA)
// ---------------------------------------------------------------------------

#define BATCH   64
#define C_IN    64
#define C_OUT   64
#define NGRID   8192
#define NMODES  64
#define MROWS   (BATCH * C_IN)       // 4096
#define JCOLS   (2 * NMODES)         // 128 (cos | sin)

typedef __attribute__((ext_vector_type(16))) _Float16  v16h;
typedef __attribute__((ext_vector_type(8)))  _Float16  v8h;
typedef __attribute__((ext_vector_type(8)))  float     v8f;
typedef __attribute__((ext_vector_type(4)))  unsigned  v4u;
typedef __attribute__((ext_vector_type(8)))  int       v8i;
typedef __attribute__((ext_vector_type(4)))  int       v4i;

#define WMMA_F16(a, b, c) \
  __builtin_amdgcn_wmma_f32_16x16x32_f16(false, (a), false, (b), (short)0, (c), false, false)

// ------------------------- workspace layout (bytes) -------------------------
#define OFF_TF_HI  (0u)
#define OFF_TF_LO  (2u  * 1024 * 1024)
#define OFF_GI_HI  (4u  * 1024 * 1024)
#define OFF_GI_LO  (6u  * 1024 * 1024)
#define OFF_XPART  (8u  * 1024 * 1024)
#define OFF_AHI    (16u * 1024 * 1024)
#define OFF_ALO    (17u * 1024 * 1024)

// =============================== stage 0 ====================================
// Fragment-ordered twiddles. Forward Tf: B of GEMM1, K=n (256 kchunks), N=j
// (8 ntiles). Inverse Gi: B of GEMM3, K=j (4 kchunks), N=n (512 ntiles).
// Layout [kchunk][ntile][lane][16 halves]: a lane's fragment = one 32B load.
__global__ __launch_bounds__(256) void twiddle_kernel(_Float16* __restrict__ tf_hi,
                                                      _Float16* __restrict__ tf_lo,
                                                      _Float16* __restrict__ gi_hi,
                                                      _Float16* __restrict__ gi_lo) {
  int t = blockIdx.x * 256 + threadIdx.x;      // 0 .. 2*2^20-1
  const int EMASK = (1 << 20) - 1;
  bool inv = t > EMASK;
  int e = t & EMASK;
  int h    = e & 15;
  int lane = (e >> 4) & 31;
  int rest = e >> 9;
  int kloc = ((lane >> 4) ? 8 : 0) + (h & 7) + ((h & 8) ? 16 : 0); // K within 32-chunk (table order)
  int n, j;
  if (!inv) {
    int ntile = rest & 7, kchunk = rest >> 3;  // K=n, N=j
    n = kchunk * 32 + kloc;
    j = ntile * 16 + (lane & 15);
  } else {
    int ntile = rest & 511, kchunk = rest >> 9; // K=j, N=n
    j = kchunk * 32 + kloc;
    n = ntile * 16 + (lane & 15);
  }
  int kmode = j & (NMODES - 1);
  int ph = (kmode * n) & (NGRID - 1);          // exact integer phase reduction
  float theta = (float)((double)ph * 7.6699039394282058e-04); // 2*pi/8192
  float val = (j < NMODES) ? cosf(theta) : sinf(theta);
  _Float16 hi = (_Float16)val;
  _Float16 lo = (_Float16)(val - (float)hi);
  if (!inv) { tf_hi[e] = hi; tf_lo[e] = lo; }
  else      { gi_hi[e] = hi; gi_lo[e] = lo; }
}

// =============================== stage 1 ====================================
#define KC      256                    // dwords staged per TDM op
#define KSPLIT  4
#define KSEG    (NGRID / KSPLIT)       // 2048
#define S1_IT   (KSEG / KC)            // 8
#define ROWD    260                    // padded f32 LDS row stride (dwords)
#define BUFD    (16 * ROWD)            // 4160 dwords per f32 buffer
#define LSTR    40                     // half-fragment per-lane stride (halves)
#define KCH     (KC / 32)              // 8 kchunks per staged tile
#define HBUF    (KCH * 32 * LSTR)      // 10240 halves per hi (or lo) buffer

__device__ __forceinline__ void tdm_load_tile(const float* gptr, unsigned lds_off) {
  uint64_t ga = (uint64_t)(uintptr_t)gptr;
  v4u g0;
  g0[0] = 1u;                                   // count=1, user descriptor
  g0[1] = lds_off;                              // lds_addr (bytes)
  g0[2] = (unsigned)(ga & 0xffffffffu);         // global_addr[31:0]
  g0[3] = (unsigned)((ga >> 32) & 0x01ffffffu) | (2u << 30);  // addr[56:32]|type=2
  v8i g1;
  g1[0] = (int)((2u << 16)      // data_size = 4B
              | (1u << 20)      // pad_enable
              | (7u << 22)      // pad_interval: 256 dwords (one row)
              | (3u << 25));    // pad_amount: 4 dwords
  unsigned td0 = NGRID;         // tensor dim0 (elements)
  unsigned td1 = 1u << 20;      // generous dim1
  g1[1] = (int)((td0 & 0xffffu) << 16);                          // dim0 lo16
  g1[2] = (int)(((td0 >> 16) & 0xffffu) | ((td1 & 0xffffu) << 16));
  g1[3] = (int)(((td1 >> 16) & 0xffffu) | ((unsigned)KC << 16)); // tile_dim0=256
  g1[4] = (int)(16u);                                            // tile_dim1=16
  g1[5] = (int)(unsigned)NGRID;                                  // dim0 stride
  g1[6] = 0;
  g1[7] = 0;
  v4i z4 = {0, 0, 0, 0};
#if defined(__clang_major__) && (__clang_major__ >= 23)
  v8i z8 = {0, 0, 0, 0, 0, 0, 0, 0};
  __builtin_amdgcn_tensor_load_to_lds(g0, g1, z4, z4, z8, 0);
#else
  __builtin_amdgcn_tensor_load_to_lds(g0, g1, z4, z4, 0);
#endif
}

// Grid: 1024 blocks = 256 row-tiles x 4 K-segments, 256 threads (8 waves).
// Per staged tile: TDM -> f32 LDS; 256 threads split once into fragment-order
// hi/lo half buffers (b128 stores); each wave then consumes pure fragments.
__global__ __launch_bounds__(256) void fwd_dft_kernel(const float* __restrict__ lhss,
                                                      const _Float16* __restrict__ tf_hi,
                                                      const _Float16* __restrict__ tf_lo,
                                                      float* __restrict__ xpart) {
  __shared__ __align__(16) float    sA[2 * BUFD];
  __shared__ __align__(16) _Float16 hA[2][HBUF];
  __shared__ __align__(16) _Float16 lA[2][HBUF];

  const int blk  = blockIdx.x;
  const int mblk = blk >> 2;             // row tile 0..255
  const int kseg = blk & 3;              // K segment
  const int wid  = threadIdx.x >> 5;     // wave id = ntile (0..7)
  const int lane = threadIdx.x & 31;
  const int r0   = mblk * 16;
  const unsigned kbase = (unsigned)kseg * KSEG;
  const unsigned lds_base = (unsigned)(uintptr_t)&sA[0];

  v8f acc = {};

  if (wid == 0)
    tdm_load_tile(lhss + (size_t)r0 * NGRID + kbase, lds_base);

  for (int it = 0; it < S1_IT; ++it) {
    const int buf = it & 1;
    if (wid == 0) __builtin_amdgcn_s_wait_tensorcnt(0);
    __syncthreads();                               // f32 tile [buf] ready
    if (wid == 0 && (it + 1) < S1_IT)
      tdm_load_tile(lhss + (size_t)r0 * NGRID + kbase + (unsigned)(it + 1) * KC,
                    lds_base + (unsigned)((buf ^ 1) * BUFD * 4));

    // ---- cooperative hi/lo split into fragment-order half buffers ----
    // 512 runs of 8 consecutive k: run g -> row m=g>>5, k8=g&31,
    // fragment position: lane L = m + (k8&1)*16, halves (k8&2 ? 8..15 : 0..7).
    const float* As = &sA[buf * BUFD];
#pragma unroll
    for (int s = 0; s < 2; ++s) {
      const int g  = threadIdx.x * 2 + s;
      const int m  = g >> 5;
      const int k8 = g & 31;
      const float* src = As + m * ROWD + k8 * 8;
      const int doff = (k8 >> 2) * (32 * LSTR)
                     + (m + ((k8 & 1) << 4)) * LSTR
                     + ((k8 & 2) ? 8 : 0);
      v8h hv, lv;
#pragma unroll
      for (int tt = 0; tt < 8; ++tt) {
        float x = src[tt];
        _Float16 hh = (_Float16)x;
        hv[tt] = hh;
        lv[tt] = (_Float16)(x - (float)hh);
      }
      *(v8h*)&hA[buf][doff] = hv;
      *(v8h*)&lA[buf][doff] = lv;
    }
    __syncthreads();                               // half fragments ready

    // ---- WMMA compute: 8 kchunks, 3-term split product ----
#pragma unroll
    for (int kc = 0; kc < KCH; ++kc) {
      v16h ah = *(const v16h*)&hA[buf][kc * (32 * LSTR) + lane * LSTR];
      v16h al = *(const v16h*)&lA[buf][kc * (32 * LSTR) + lane * LSTR];
      const int kcg = kseg * (KSEG / 32) + it * KCH + kc;   // global kchunk
      const size_t boff = ((((size_t)kcg * 8) + wid) * 32 + lane) * 16;
      v16h bh = *(const v16h*)(tf_hi + boff);
      v16h bl = *(const v16h*)(tf_lo + boff);
      acc = WMMA_F16(ah, bh, acc);   // hi*hi
      acc = WMMA_F16(ah, bl, acc);   // hi*lo
      acc = WMMA_F16(al, bh, acc);   // lo*hi
    }
  }

  // D layout: N = lane&15, vgpr v holds M = v + (lane>=16 ? 8 : 0)
  float* xp = xpart + (size_t)kseg * MROWS * JCOLS;
  const int col   = wid * 16 + (lane & 15);
  const int rbase = r0 + ((lane >> 4) ? 8 : 0);
#pragma unroll
  for (int v = 0; v < 8; ++v)
    xp[(size_t)(rbase + v) * JCOLS + col] = acc[v];
}

// =============================== stage 2 ====================================
// Exact f32 complex mixing + irfft coefficient packing, written hi/lo-split
// directly in WMMA A-fragment order for stage 3. Deterministic (fixed-order
// K-segment reduction, no FP atomics).
__device__ __forceinline__ void store_split(_Float16* __restrict__ ahi,
                                            _Float16* __restrict__ alo,
                                            int r, int j, float v) {
  int mtile = r >> 4, m = r & 15;
  int kc = j >> 5, kk = j & 31;
  int L = m + (((kk >> 3) & 1) ? 16 : 0);
  int h = (kk & 7) + ((kk & 16) ? 8 : 0);
  size_t idx = ((((size_t)mtile * 4) + kc) * 32 + L) * 16 + h;
  _Float16 hi = (_Float16)v;
  ahi[idx] = hi;
  alo[idx] = (_Float16)(v - (float)hi);
}

__global__ __launch_bounds__(256) void mix_kernel(const float* __restrict__ xpart,
                                                  const float* __restrict__ w_real,
                                                  const float* __restrict__ w_imag,
                                                  _Float16* __restrict__ ahi,
                                                  _Float16* __restrict__ alo) {
  int t = blockIdx.x * 256 + threadIdx.x;  // 0..262143
  int k = t & (NMODES - 1);
  int r = t >> 6;                          // b*C_OUT + o
  int o = r & (C_OUT - 1);
  int b = r >> 6;
  float mr = 0.f, mi = 0.f;
  for (int i = 0; i < C_IN; ++i) {
    int ri = b * C_IN + i;
    float xr = 0.f, xs = 0.f;
#pragma unroll
    for (int p = 0; p < KSPLIT; ++p) {
      const float* xp = xpart + ((size_t)p * MROWS + ri) * JCOLS;
      xr += xp[k];
      xs += xp[NMODES + k];
    }
    float xi = -xs;                        // rfft: X = sum x cos - i sum x sin
    float wr = w_real[((size_t)i * C_OUT + o) * NMODES + k];
    float wi = w_imag[((size_t)i * C_OUT + o) * NMODES + k];
    mr += xr * wr - xi * wi;
    mi += xr * wi + xi * wr;
  }
  // irfft with only modes 0..63: out = (1/N)[Re M0 + 2*sum_k (Re cos - Im sin)]
  const float invn = 1.0f / (float)NGRID;
  float scale = (k == 0) ? invn : 2.0f * invn;
  float cr = scale * mr;                        // multiplies cos(2πkn/N)
  float ci = (k == 0) ? 0.0f : (-scale * mi);   // multiplies sin(2πkn/N)
  store_split(ahi, alo, r, k, cr);
  store_split(ahi, alo, r, NMODES + k, ci);
}

// =============================== stage 3 ====================================
// Inverse GEMM: out[r][n] = sum_j coef[r][j] * g(j,n).  M=4096, N=8192, K=128.
// All operands pre-packed in fragment order -> pure coalesced loads + WMMA.
__global__ __launch_bounds__(256) void inv_dft_kernel(const _Float16* __restrict__ ahi,
                                                      const _Float16* __restrict__ alo,
                                                      const _Float16* __restrict__ gi_hi,
                                                      const _Float16* __restrict__ gi_lo,
                                                      float* __restrict__ out) {
  const int blk   = blockIdx.x;
  const int mtile = blk >> 6;               // 0..255
  const int ng    = blk & 63;               // n-group of 8 tiles
  const int wid   = threadIdx.x >> 5;
  const int lane  = threadIdx.x & 31;
  const int ntile = ng * 8 + wid;           // 0..511

  v8f acc = {};
#pragma unroll
  for (int kc = 0; kc < JCOLS / 32; ++kc) {
    size_t aoff = ((((size_t)mtile * 4) + kc) * 32 + lane) * 16;
    v16h ah = *(const v16h*)(ahi + aoff);
    v16h al = *(const v16h*)(alo + aoff);
    size_t boff = ((((size_t)kc * 512) + ntile) * 32 + lane) * 16;
    v16h bh = *(const v16h*)(gi_hi + boff);
    v16h bl = *(const v16h*)(gi_lo + boff);
    acc = WMMA_F16(ah, bh, acc);
    acc = WMMA_F16(ah, bl, acc);
    acc = WMMA_F16(al, bh, acc);
  }
  const int col  = ntile * 16 + (lane & 15);
  const int row0 = mtile * 16 + ((lane >> 4) ? 8 : 0);
#pragma unroll
  for (int v = 0; v < 8; ++v)
    out[(size_t)(row0 + v) * NGRID + col] = acc[v];
}

// ================================ launch ====================================
extern "C" void kernel_launch(void* const* d_in, const int* in_sizes, int n_in,
                              void* d_out, int out_size, void* d_ws, size_t ws_size,
                              hipStream_t stream) {
  const float* lhss   = (const float*)d_in[0];
  const float* w_real = (const float*)d_in[1];
  const float* w_imag = (const float*)d_in[2];
  float* out = (float*)d_out;
  char*  ws  = (char*)d_ws;

  _Float16* tf_hi = (_Float16*)(ws + OFF_TF_HI);
  _Float16* tf_lo = (_Float16*)(ws + OFF_TF_LO);
  _Float16* gi_hi = (_Float16*)(ws + OFF_GI_HI);
  _Float16* gi_lo = (_Float16*)(ws + OFF_GI_LO);
  float*    xpart = (float*)   (ws + OFF_XPART);
  _Float16* ahi   = (_Float16*)(ws + OFF_AHI);
  _Float16* alo   = (_Float16*)(ws + OFF_ALO);

  twiddle_kernel<<<8192, 256, 0, stream>>>(tf_hi, tf_lo, gi_hi, gi_lo);
  fwd_dft_kernel<<<(MROWS / 16) * KSPLIT, 256, 0, stream>>>(lhss, tf_hi, tf_lo, xpart);
  mix_kernel<<<(MROWS * NMODES) / 256, 256, 0, stream>>>(xpart, w_real, w_imag, ahi, alo);
  inv_dft_kernel<<<(MROWS / 16) * (NGRID / 128), 256, 0, stream>>>(ahi, alo, gi_hi, gi_lo, out);
}
